// GSA_26834955665772
// MI455X (gfx1250) — matrix operand order repaired and checked by reading, not compiled, not run on hardware
//
#include <hip/hip_runtime.h>
#include <hip/hip_bf16.h>
#include <math.h>

// ---------------------------------------------------------------------------
// Problem-size constants (fixed by the reference setup_inputs()).
// ---------------------------------------------------------------------------
#define B_  4
#define N_  1024
#define D_  1024
#define F_  2048
#define M_  (B_ * N_)     // 4096 token rows
#define NH_ 8
#define KD_ 128           // head dim
#define HP_ 32
#define WP_ 32

typedef __attribute__((ext_vector_type(16))) _Float16 v16h;
typedef __attribute__((ext_vector_type(8)))  float    v8f;

static __device__ __forceinline__ float gelu_exact(float x) {
  return 0.5f * x * (1.0f + erff(x * 0.70710678118654752440f));
}

// Raw LDS byte address for async global->LDS DMA (addr[31:0] of the generic
// pointer is the LDS offset after the aperture check, ISA 10.2).
static __device__ __forceinline__ unsigned lds_addr(const void* p) {
  return (unsigned)(unsigned long long)p;
}

// ---------------------------------------------------------------------------
// Depth map 8x8 mean-pool:  (B,256,256) -> (B,1024)
// ---------------------------------------------------------------------------
__global__ void gsa_pool_kernel(const float* __restrict__ dm, float* __restrict__ dtok) {
  int idx = blockIdx.x * blockDim.x + threadIdx.x;
  if (idx >= B_ * N_) return;
  int n = idx & (N_ - 1);
  int b = idx >> 10;
  int ti = n >> 5, tj = n & 31;
  float s = 0.f;
  for (int i = 0; i < 8; ++i)
    for (int j = 0; j < 8; ++j)
      s += dm[((size_t)b * 256 + ti * 8 + i) * 256 + tj * 8 + j];
  dtok[idx] = s * (1.f / 64.f);
}

// ---------------------------------------------------------------------------
// Depthwise conv (KHxKH, pad), NHWC, optional residual add of the input.
// ---------------------------------------------------------------------------
__global__ void gsa_dwconv_kernel(const float* __restrict__ in, const float* __restrict__ w,
                                  const float* __restrict__ bias, float* __restrict__ out,
                                  int C, int KH, int pad, int add_res) {
  int c4n = C >> 2;
  long long idx = (long long)blockIdx.x * blockDim.x + threadIdx.x;
  long long total = (long long)B_ * HP_ * WP_ * c4n;
  if (idx >= total) return;
  int c4 = (int)(idx % c4n) << 2;
  long long t = idx / c4n;
  int x = (int)(t % WP_); t /= WP_;
  int y = (int)(t % HP_); t /= HP_;
  int b = (int)t;
  float4 acc; acc.x = acc.y = acc.z = acc.w = 0.f;
  for (int ky = 0; ky < KH; ++ky) {
    int iy = y + ky - pad;
    if (iy < 0 || iy >= HP_) continue;
    for (int kx = 0; kx < KH; ++kx) {
      int ix = x + kx - pad;
      if (ix < 0 || ix >= WP_) continue;
      const float4 xv = *(const float4*)&in[(((size_t)b * HP_ + iy) * WP_ + ix) * C + c4];
      const float4 wv = *(const float4*)&w[((size_t)ky * KH + kx) * C + c4];
      acc.x += xv.x * wv.x; acc.y += xv.y * wv.y;
      acc.z += xv.z * wv.z; acc.w += xv.w * wv.w;
    }
  }
  const float4 bv = *(const float4*)&bias[c4];
  acc.x += bv.x; acc.y += bv.y; acc.z += bv.z; acc.w += bv.w;
  if (add_res) {
    const float4 cv = *(const float4*)&in[(((size_t)b * HP_ + y) * WP_ + x) * C + c4];
    acc.x += cv.x; acc.y += cv.y; acc.z += cv.z; acc.w += cv.w;
  }
  *(float4*)&out[(((size_t)b * HP_ + y) * WP_ + x) * C + c4] = acc;
}

// ---------------------------------------------------------------------------
// LayerNorm over D_ (one workgroup per row), output f16 for the WMMA GEMMs.
// ---------------------------------------------------------------------------
__global__ __launch_bounds__(256)
void gsa_ln_kernel(const float* __restrict__ x, const float* __restrict__ g,
                   const float* __restrict__ bb, _Float16* __restrict__ out) {
  __shared__ float red[256];
  const int row = blockIdx.x, tid = threadIdx.x;
  const float* xr = x + (size_t)row * D_;
  float s = 0.f;
  for (int i = tid; i < D_; i += 256) s += xr[i];
  red[tid] = s; __syncthreads();
  for (int st = 128; st; st >>= 1) { if (tid < st) red[tid] += red[tid + st]; __syncthreads(); }
  const float mean = red[0] * (1.f / D_);
  __syncthreads();
  float v = 0.f;
  for (int i = tid; i < D_; i += 256) { float d = xr[i] - mean; v += d * d; }
  red[tid] = v; __syncthreads();
  for (int st = 128; st; st >>= 1) { if (tid < st) red[tid] += red[tid + st]; __syncthreads(); }
  const float rstd = rsqrtf(red[0] * (1.f / D_) + 1e-5f);
  for (int i = tid; i < D_; i += 256)
    out[(size_t)row * D_ + i] = (_Float16)((xr[i] - mean) * rstd * g[i] + bb[i]);
}

// ---------------------------------------------------------------------------
// Weight convert+transpose:  W[K][N] f32  ->  Wt[N][K] f16
// ---------------------------------------------------------------------------
__global__ void gsa_wtrans_kernel(const float* __restrict__ W, _Float16* __restrict__ Wt,
                                  int K, int Nc) {
  long long idx = (long long)blockIdx.x * blockDim.x + threadIdx.x;
  if (idx >= (long long)K * Nc) return;
  int n = (int)(idx % Nc);
  int k = (int)(idx / Nc);
  Wt[(size_t)n * K + k] = (_Float16)W[idx];
}

__global__ void gsa_f2h_kernel(const float* __restrict__ a, _Float16* __restrict__ o, long long n) {
  long long i = (long long)blockIdx.x * blockDim.x + threadIdx.x;
  if (i < n) o[i] = (_Float16)a[i];
}

__global__ void gsa_addcvt_kernel(const float* __restrict__ a, const float* __restrict__ b,
                                  _Float16* __restrict__ o, long long n) {
  long long i = (long long)blockIdx.x * blockDim.x + threadIdx.x;
  if (i < n) o[i] = (_Float16)(a[i] + b[i]);
}

// ---------------------------------------------------------------------------
// RoPE + (B,N,nh,kd) -> (B,nh,N,kd) pack to f16.   do_rope=0 for V.
// ---------------------------------------------------------------------------
__global__ void gsa_rope_pack_kernel(const float* __restrict__ in, _Float16* __restrict__ outp,
                                     int do_rope) {
  long long idx = (long long)blockIdx.x * blockDim.x + threadIdx.x;
  long long total = (long long)M_ * (D_ / 2);
  if (idx >= total) return;
  int p = (int)(idx % (KD_ / 2));
  long long t = idx / (KD_ / 2);
  int h = (int)(t % NH_); t /= NH_;
  int n = (int)(t % N_);
  int b = (int)(t / N_);
  size_t src = ((size_t)(b * N_ + n)) * D_ + h * KD_ + 2 * p;
  size_t dst = (((size_t)(b * NH_ + h)) * N_ + n) * KD_ + 2 * p;
  float x0 = in[src], x1 = in[src + 1];
  if (do_rope) {
    float a  = powf(10000.f, -(float)p * (1.f / 63.f));   // linspace(0,1,64), step 1/63
    float th = (float)n * a;
    float sn = sinf(th), cs = cosf(th);
    float o0 = x0 * cs + x1 * sn;
    float o1 = x1 * cs - x0 * sn;
    x0 = o0; x1 = o1;
  }
  outp[dst] = (_Float16)x0;
  outp[dst + 1] = (_Float16)x1;
}

// ---------------------------------------------------------------------------
// WMMA GEMM:  out[M_][Nc] f32 = act( (A[M_][K]f16 @ Wt[Nc][K]f16^T + bias)*scale ) + resid
// 128x128 tile / 256 threads (8 waves, 2x4), BK=32 -> v_wmma_f32_16x16x32_f16.
// Double-buffered LDS; tiles staged with GLOBAL_LOAD_ASYNC_TO_LDS_B128 so the
// next tile's DMA overlaps the current tile's WMMAs (ASYNCcnt, one barrier/step).
// ---------------------------------------------------------------------------
__global__ __launch_bounds__(256)
void gsa_gemm_kernel(const _Float16* __restrict__ A, const _Float16* __restrict__ Wt,
                     const float* __restrict__ bias, const float* __restrict__ resid,
                     float* __restrict__ out, int K, int Nc, float scale, int act) {
  __shared__ _Float16 Al[2][128 * 40];
  __shared__ _Float16 Bl[2][128 * 40];
  const int tid = threadIdx.x, lane = tid & 31, wid = tid >> 5;
  const int half = lane >> 4, l16 = lane & 15;
  const int bm = blockIdx.y * 128, bn = blockIdx.x * 128;
  const int wm = wid & 1, wn = wid >> 1;
  const int m0 = wm * 64, n0 = wn * 32;
  const int srow = tid >> 1, sh = tid & 1;

  v8f acc[4][2] = {};

  // Async global->LDS staging: each thread DMAs 32B of A and 32B of Wt.
  // INST_OFFSET applies to both the LDS and global address (ISA 08 §4.4).
  auto stage = [&](int buf, int k0) {
    const _Float16* gA = &A[(size_t)(bm + srow) * K + k0 + sh * 16];
    const _Float16* gB = &Wt[(size_t)(bn + srow) * K + k0 + sh * 16];
    unsigned lA = lds_addr(&Al[buf][srow * 40 + sh * 16]);
    unsigned lB = lds_addr(&Bl[buf][srow * 40 + sh * 16]);
    asm volatile(
        "global_load_async_to_lds_b128 %0, %2, off\n\t"
        "global_load_async_to_lds_b128 %0, %2, off offset:16\n\t"
        "global_load_async_to_lds_b128 %1, %3, off\n\t"
        "global_load_async_to_lds_b128 %1, %3, off offset:16"
        :: "v"(lA), "v"(lB), "v"(gA), "v"(gB) : "memory");
  };

  stage(0, 0);
  asm volatile("s_wait_asynccnt 0" ::: "memory");
  __syncthreads();

  int cur = 0;
  for (int k0 = 0; k0 < K; k0 += 32) {
    if (k0 + 32 < K) stage(cur ^ 1, k0 + 32);  // overlap DMA with WMMA

    union { v16h v; unsigned u[8]; } af[4];
    union { v16h v; uint4 q[2]; } bf[2];
#pragma unroll
    for (int i = 0; i < 4; ++i) {
      const int mrow = m0 + i * 16 + l16;
#pragma unroll
      for (int vv = 0; vv < 8; ++vv) {
        // A-matrix 16x32 layout (contiguous K, matching B): VGPR v holds K pair
        // {2v,2v+1}+16*half.
        const int kk = (half << 4) + (vv << 1);
        af[i].u[vv] = *(const unsigned*)&Al[cur][mrow * 40 + kk];
      }
    }
#pragma unroll
    for (int j = 0; j < 2; ++j) {
      const int ncol = n0 + j * 16 + l16;
      const uint4* p = (const uint4*)&Bl[cur][ncol * 40 + (half << 4)];
      bf[j].q[0] = p[0]; bf[j].q[1] = p[1];   // B: lane=col, K=half*16+j contiguous
    }
#pragma unroll
    for (int i = 0; i < 4; ++i)
#pragma unroll
      for (int j = 0; j < 2; ++j)
        acc[i][j] = __builtin_amdgcn_wmma_f32_16x16x32_f16(
            false, af[i].v, false, bf[j].v, (short)0, acc[i][j], false, false);

    asm volatile("s_wait_asynccnt 0" ::: "memory");
    __syncthreads();
    cur ^= 1;
  }

  // Epilogue: C layout — VGPR r holds row (r + 8*half), col = lane&15.
#pragma unroll
  for (int i = 0; i < 4; ++i)
#pragma unroll
    for (int j = 0; j < 2; ++j)
#pragma unroll
      for (int r = 0; r < 8; ++r) {
        const int row = bm + m0 + i * 16 + half * 8 + r;
        const int col = bn + n0 + j * 16 + l16;
        float val = acc[i][j][r] + bias[col];
        val *= scale;
        if (act) val = gelu_exact(val);
        if (resid) val += resid[(size_t)row * Nc + col];
        out[(size_t)row * Nc + col] = val;
      }
}

// ---------------------------------------------------------------------------
// Fused masked flash-attention (per b,h).  128 threads = 4 waves, each wave
// owns 16 query rows; K tile DMA'd to LDS via async-to-LDS, V tile transposed
// through VGPRs.  S = QK^T via WMMA, position+depth decay mask added in the
// C-register layout, online softmax with 16-lane shfl reductions, P transposed
// through a wave-private LDS patch (s_wait_dscnt), O += P@V via WMMA.
// ---------------------------------------------------------------------------
__global__ __launch_bounds__(128)
void gsa_attn_kernel(const _Float16* __restrict__ qf, const _Float16* __restrict__ kf,
                     const _Float16* __restrict__ vf, const float* __restrict__ dtok,
                     const float* __restrict__ geo, float* __restrict__ outp) {
  __shared__ _Float16 Kl[32 * 136];    // K tile, 136-hw stride (16B aligned rows)
  __shared__ _Float16 Vt[128 * 40];    // V tile transposed: [out-dim][kv]
  __shared__ _Float16 Pl[4][16 * 32];  // per-wave P transpose patch

  const int lane = threadIdx.x & 31, wid = threadIdx.x >> 5;
  const int half = lane >> 4, l16 = lane & 15;
  const int bh = blockIdx.y, b = bh >> 3, h = bh & 7;
  const int q0 = blockIdx.x * 64 + wid * 16;
  const float g0 = geo[0], g1 = geo[1];
  const float dec = logf(1.f - exp2f(-2.f - 0.5f * (float)h));  // log(1-2^-(2+h/2))

  // Q fragments (16 rows x 128 kd) -> 4 A-fragments, kept in VGPRs.
  union AFu { v16h v; unsigned u[8]; };
  AFu qa[4];
  const size_t qbase = ((size_t)bh * N_ + q0 + l16) * KD_;
#pragma unroll
  for (int kc = 0; kc < 4; ++kc)
#pragma unroll
    for (int vv = 0; vv < 8; ++vv) {
      const int kk = (half << 4) + (vv << 1);
      qa[kc].u[vv] = *(const unsigned*)&qf[qbase + kc * 32 + kk];
    }

  float dq[8]; int qh[8], qw[8];
#pragma unroll
  for (int r = 0; r < 8; ++r) {
    const int qn = q0 + half * 8 + r;
    dq[r] = dtok[b * N_ + qn];
    qh[r] = qn >> 5; qw[r] = qn & 31;
  }

  v8f o[8] = {};
  float mrun[8], lrun[8];
#pragma unroll
  for (int r = 0; r < 8; ++r) { mrun[r] = -1e30f; lrun[r] = 0.f; }

  const int srow = threadIdx.x >> 2, sq = threadIdx.x & 3;  // staging map

  for (int kv0 = 0; kv0 < N_; kv0 += 32) {
    __syncthreads();
    // Stage K tile (32x128): 64B per thread via async global->LDS DMA.
    {
      const _Float16* gk = &kf[((size_t)bh * N_ + kv0 + srow) * KD_ + sq * 32];
      unsigned lk = lds_addr(&Kl[srow * 136 + sq * 32]);
      asm volatile(
          "global_load_async_to_lds_b128 %0, %1, off\n\t"
          "global_load_async_to_lds_b128 %0, %1, off offset:16\n\t"
          "global_load_async_to_lds_b128 %0, %1, off offset:32\n\t"
          "global_load_async_to_lds_b128 %0, %1, off offset:48"
          :: "v"(lk), "v"(gk) : "memory");
    }
    // Stage V tile transposed: Vt[out-dim][kv-row]  (needs the VGPR path).
    const _Float16* gv = &vf[((size_t)bh * N_ + kv0 + srow) * KD_ + sq * 32];
#pragma unroll
    for (int e = 0; e < 32; ++e) Vt[(sq * 32 + e) * 40 + srow] = gv[e];
    asm volatile("s_wait_asynccnt 0" ::: "memory");
    __syncthreads();

    // S = Q @ K^T  (two 16x16 tiles along kv)
    v8f s[2] = {};
#pragma unroll
    for (int t16 = 0; t16 < 2; ++t16)
#pragma unroll
      for (int kc = 0; kc < 4; ++kc) {
        union { v16h v; uint4 q[2]; } kb;
        const uint4* p = (const uint4*)&Kl[(t16 * 16 + l16) * 136 + kc * 32 + (half << 4)];
        kb.q[0] = p[0]; kb.q[1] = p[1];
        s[t16] = __builtin_amdgcn_wmma_f32_16x16x32_f16(
            false, qa[kc].v, false, kb.v, (short)0, s[t16], false, false);
      }

    // Mask + online softmax (rows live across the 16-lane half-group).
#pragma unroll
    for (int r = 0; r < 8; ++r) {
#pragma unroll
      for (int t16 = 0; t16 < 2; ++t16) {
        const int kn = kv0 + t16 * 16 + l16;
        const float manh = fabsf((float)(qh[r] - (kn >> 5))) + fabsf((float)(qw[r] - (kn & 31)));
        const float dd = fabsf(dq[r] - dtok[b * N_ + kn]);
        s[t16][r] += (g0 * manh + g1 * dd) * dec;
      }
      float rm = fmaxf(s[0][r], s[1][r]);
#pragma unroll
      for (int mH = 1; mH < 16; mH <<= 1) rm = fmaxf(rm, __shfl_xor(rm, mH, 32));
      const float mn = fmaxf(mrun[r], rm);
      const float sc = expf(mrun[r] - mn);
      mrun[r] = mn;
      const float p0 = expf(s[0][r] - mn);
      const float p1 = expf(s[1][r] - mn);
      float rs = p0 + p1;
#pragma unroll
      for (int mH = 1; mH < 16; mH <<= 1) rs += __shfl_xor(rs, mH, 32);
      lrun[r] = lrun[r] * sc + rs;
#pragma unroll
      for (int c = 0; c < 8; ++c) o[c][r] = o[c][r] * sc;
      // Write P (C layout) into wave-private LDS patch for transpose.
      const int prow = half * 8 + r;
      Pl[wid][prow * 32 + l16] = (_Float16)p0;
      Pl[wid][prow * 32 + 16 + l16] = (_Float16)p1;
    }
    // Wave-private LDS RAW: drain DS counter explicitly (CDNA5 split counters).
    asm volatile("s_wait_dscnt 0" ::: "memory");

    // Re-read P as a 16x32 A-fragment.
    AFu pa;
#pragma unroll
    for (int vv = 0; vv < 8; ++vv) {
      const int kk = (half << 4) + (vv << 1);
      pa.u[vv] = *(const unsigned*)&Pl[wid][l16 * 32 + kk];
    }
    // O += P @ V  (8 output chunks of 16 along kd)
#pragma unroll
    for (int c = 0; c < 8; ++c) {
      union { v16h v; uint4 q[2]; } vb_;
      const uint4* pv = (const uint4*)&Vt[(c * 16 + l16) * 40 + (half << 4)];
      vb_.q[0] = pv[0]; vb_.q[1] = pv[1];
      o[c] = __builtin_amdgcn_wmma_f32_16x16x32_f16(
          false, pa.v, false, vb_.v, (short)0, o[c], false, false);
    }
  }

  // Normalize and scatter back to (B,N,D) layout.
#pragma unroll
  for (int c = 0; c < 8; ++c)
#pragma unroll
    for (int r = 0; r < 8; ++r) {
      const int qn = q0 + half * 8 + r;
      const int col = h * KD_ + c * 16 + l16;
      outp[((size_t)b * N_ + qn) * D_ + col] = o[c][r] / lrun[r];
    }
}

// ---------------------------------------------------------------------------
// Driver
// ---------------------------------------------------------------------------
extern "C" void kernel_launch(void* const* d_in, const int* in_sizes, int n_in,
                              void* d_out, int out_size, void* d_ws, size_t ws_size,
                              hipStream_t stream) {
  const float* vt    = (const float*)d_in[0];
  const float* dm    = (const float*)d_in[1];
  const float* geo   = (const float*)d_in[2];
  const float* cnn_w = (const float*)d_in[3];
  const float* cnn_b = (const float*)d_in[4];
  const float* ln1g  = (const float*)d_in[5];
  const float* ln1b  = (const float*)d_in[6];
  const float* qw    = (const float*)d_in[7];
  const float* qb    = (const float*)d_in[8];
  const float* kw    = (const float*)d_in[9];
  const float* kb    = (const float*)d_in[10];
  const float* vw    = (const float*)d_in[11];
  const float* vb    = (const float*)d_in[12];
  const float* lepew = (const float*)d_in[13];
  const float* lepeb = (const float*)d_in[14];
  const float* ow    = (const float*)d_in[15];
  const float* ob    = (const float*)d_in[16];
  const float* ln2g  = (const float*)d_in[17];
  const float* ln2b  = (const float*)d_in[18];
  const float* fc1w  = (const float*)d_in[19];
  const float* fc1b  = (const float*)d_in[20];
  const float* dww   = (const float*)d_in[21];
  const float* dwb   = (const float*)d_in[22];
  const float* fc2w  = (const float*)d_in[23];
  const float* fc2b  = (const float*)d_in[24];
  float* outp = (float*)d_out;

  char* wsp = (char*)d_ws;
  size_t off = 0;
  auto alloc = [&](size_t bytes) -> void* {
    void* p = wsp + off;
    off = (off + bytes + 255) & ~(size_t)255;
    return p;
  };

  float*    xA    = (float*)alloc((size_t)M_ * D_ * 4);
  float*    xB    = (float*)alloc((size_t)M_ * D_ * 4);
  _Float16* xn    = (_Float16*)alloc((size_t)M_ * D_ * 2);
  float*    qf32  = (float*)alloc((size_t)M_ * D_ * 4);   // reused as h1 (with kf32)
  float*    kf32  = (float*)alloc((size_t)M_ * D_ * 4);
  float*    vf32  = (float*)alloc((size_t)M_ * D_ * 4);
  float*    lepe  = (float*)alloc((size_t)M_ * D_ * 4);   // reused as h2 (with attn)
  float*    attn  = (float*)alloc((size_t)M_ * D_ * 4);
  float*    dtok  = (float*)alloc((size_t)B_ * N_ * 4);
  _Float16* qfh   = (_Float16*)alloc((size_t)M_ * D_ * 2);
  _Float16* kfh   = (_Float16*)alloc((size_t)M_ * D_ * 2);
  _Float16* vfh   = (_Float16*)alloc((size_t)M_ * D_ * 2);
  _Float16* aoutH = (_Float16*)alloc((size_t)M_ * D_ * 2);
  _Float16* h1H   = (_Float16*)alloc((size_t)M_ * F_ * 2);
  _Float16* qwT   = (_Float16*)alloc((size_t)D_ * D_ * 2);
  _Float16* kwT   = (_Float16*)alloc((size_t)D_ * D_ * 2);
  _Float16* vwT   = (_Float16*)alloc((size_t)D_ * D_ * 2);
  _Float16* owT   = (_Float16*)alloc((size_t)D_ * D_ * 2);
  _Float16* fc1T  = (_Float16*)alloc((size_t)D_ * F_ * 2);
  _Float16* fc2T  = (_Float16*)alloc((size_t)F_ * D_ * 2);
  float* h1 = qf32;   // 32 MB span over qf32+kf32 (dead after RoPE pack)
  float* h2 = lepe;   // 32 MB span over lepe+attn  (dead after o-proj)
  (void)in_sizes; (void)n_in; (void)out_size; (void)ws_size;

  const float kscale = 1.f / sqrtf((float)KD_);
  const long long eMD = (long long)M_ * D_;
  const long long eMF = (long long)M_ * F_;

  // depth pooling (block-independent)
  gsa_pool_kernel<<<(B_ * N_ + 255) / 256, 256, 0, stream>>>(dm, dtok);

  for (int blk = 0; blk < 2; ++blk) {
    const float* xin  = (blk == 0) ? vt : xA;
    float*       xcur = (blk == 0) ? xA : xB;
    float*       xfin = (blk == 0) ? xA : outp;   // last MLP GEMM writes d_out

    const int gD4 = (B_ * HP_ * WP_ * (D_ / 4) + 255) / 256;
    const int gF4 = (B_ * HP_ * WP_ * (F_ / 4) + 255) / 256;
    dim3 gemmD(D_ / 128, M_ / 128), gemmF(F_ / 128, M_ / 128);

    // 1) x = x + dwconv3(x)
    gsa_dwconv_kernel<<<gD4, 256, 0, stream>>>(xin, cnn_w + (size_t)blk * 9 * D_,
                                               cnn_b + (size_t)blk * D_, xcur, D_, 3, 1, 1);
    // 2) xn = LN1(x) -> f16
    gsa_ln_kernel<<<M_, 256, 0, stream>>>(xcur, ln1g + (size_t)blk * D_, ln1b + (size_t)blk * D_, xn);
    // 3) Q/K/V projections (WMMA GEMMs)
    gsa_wtrans_kernel<<<(D_ * D_ + 255) / 256, 256, 0, stream>>>(qw + (size_t)blk * D_ * D_, qwT, D_, D_);
    gsa_wtrans_kernel<<<(D_ * D_ + 255) / 256, 256, 0, stream>>>(kw + (size_t)blk * D_ * D_, kwT, D_, D_);
    gsa_wtrans_kernel<<<(D_ * D_ + 255) / 256, 256, 0, stream>>>(vw + (size_t)blk * D_ * D_, vwT, D_, D_);
    gsa_gemm_kernel<<<gemmD, 256, 0, stream>>>(xn, qwT, qb + (size_t)blk * D_, nullptr, qf32, D_, D_, 1.f, 0);
    gsa_gemm_kernel<<<gemmD, 256, 0, stream>>>(xn, kwT, kb + (size_t)blk * D_, nullptr, kf32, D_, D_, kscale, 0);
    gsa_gemm_kernel<<<gemmD, 256, 0, stream>>>(xn, vwT, vb + (size_t)blk * D_, nullptr, vf32, D_, D_, 1.f, 0);
    // 4) LePE = dwconv5(v)
    gsa_dwconv_kernel<<<gD4, 256, 0, stream>>>(vf32, lepew + (size_t)blk * 25 * D_,
                                               lepeb + (size_t)blk * D_, lepe, D_, 5, 2, 0);
    // 5) RoPE + head pack to f16
    const int gP = (int)((eMD / 2 + 255) / 256);
    gsa_rope_pack_kernel<<<gP, 256, 0, stream>>>(qf32, qfh, 1);
    gsa_rope_pack_kernel<<<gP, 256, 0, stream>>>(kf32, kfh, 1);
    gsa_rope_pack_kernel<<<gP, 256, 0, stream>>>(vf32, vfh, 0);
    // 6) fused masked flash attention
    gsa_attn_kernel<<<dim3(N_ / 64, B_ * NH_), 128, 0, stream>>>(qfh, kfh, vfh, dtok,
                                                                 geo + (size_t)blk * 2, attn);
    // 7) (attn + lepe) -> f16, then o-proj with residual (in place on xcur)
    gsa_addcvt_kernel<<<(int)((eMD + 255) / 256), 256, 0, stream>>>(attn, lepe, aoutH, eMD);
    gsa_wtrans_kernel<<<(D_ * D_ + 255) / 256, 256, 0, stream>>>(ow + (size_t)blk * D_ * D_, owT, D_, D_);
    gsa_gemm_kernel<<<gemmD, 256, 0, stream>>>(aoutH, owT, ob + (size_t)blk * D_, xcur, xcur, D_, D_, 1.f, 0);
    // 8) MLP: h1 = gelu(LN2(x) @ fc1 + b); h2 = h1 + dwconv3(h1); x = x + h2 @ fc2 + b
    gsa_ln_kernel<<<M_, 256, 0, stream>>>(xcur, ln2g + (size_t)blk * D_, ln2b + (size_t)blk * D_, xn);
    gsa_wtrans_kernel<<<(D_ * F_ + 255) / 256, 256, 0, stream>>>(fc1w + (size_t)blk * D_ * F_, fc1T, D_, F_);
    gsa_gemm_kernel<<<gemmF, 256, 0, stream>>>(xn, fc1T, fc1b + (size_t)blk * F_, nullptr, h1, D_, F_, 1.f, 1);
    gsa_dwconv_kernel<<<gF4, 256, 0, stream>>>(h1, dww + (size_t)blk * 9 * F_,
                                               dwb + (size_t)blk * F_, h2, F_, 3, 1, 1);
    gsa_f2h_kernel<<<(int)((eMF + 255) / 256), 256, 0, stream>>>(h2, h1H, eMF);
    gsa_wtrans_kernel<<<(F_ * D_ + 255) / 256, 256, 0, stream>>>(fc2w + (size_t)blk * F_ * D_, fc2T, F_, D_);
    gsa_gemm_kernel<<<gemmD, 256, 0, stream>>>(h1H, fc2T, fc2b + (size_t)blk * D_, xcur, xfin, F_, D_, 1.f, 0);
  }
}
